// ClusteredAttention_86071144612555
// MI455X (gfx1250) — matrix operand order repaired and checked by reading, not compiled, not run on hardware
//
#include <hip/hip_runtime.h>
#include <hip/hip_bf16.h>

// ---------------------------------------------------------------------------
// Problem constants (match reference)
// ---------------------------------------------------------------------------
#define NB    4
#define LQ    4096
#define SK    4096
#define HH    16
#define EE    64
#define BITS  32
#define CL    256
#define ITERS 10
#define NH    (NB * HH)          // 64

#define PADKV 69                 // f32 LDS row pad (V staging, conflict-free)
#define KST   35                 // dword stride for bf16-pair images (K, Vt)
#define PBST  70                 // ushort stride for P buffer (even -> dword reads)

typedef __attribute__((ext_vector_type(16))) __bf16 v16bf;
typedef __attribute__((ext_vector_type(8)))  float  v8f;

union BF16x16 { v16bf v; unsigned short u[16]; unsigned d[8]; };

__device__ __forceinline__ unsigned short f2bf(float f) {
  union { __bf16 b; unsigned short u; } t;
  t.b = (__bf16)f;                                  // RNE, native cvt
  return t.u;
}

__device__ __forceinline__ unsigned pk_bf16(float lo, float hi) {
  union { __bf16 b[2]; unsigned u; } t;
  t.b[0] = (__bf16)lo;                              // low 16 bits = first elem
  t.b[1] = (__bf16)hi;
  return t.u;
}

__device__ __forceinline__ v8f wmma_bf16f32(v16bf a, v16bf b, v8f c) {
  // D = A(16x32 bf16) * B(32x16 bf16) + C(16x16 f32)
  return __builtin_amdgcn_wmma_f32_16x16x32_bf16(false, a, false, b, (short)0, c,
                                                 false, false);
}

// ---------------------------------------------------------------------------
// K1: LSH bit packing.  bit b of bits[nh][l] = (q[l] . planes[b,:E] + bias > 0)
// Also seeds centroids: cent[nh][c] = bits[nh][c] for c < 256.
// ---------------------------------------------------------------------------
__global__ __launch_bounds__(256) void k_packbits(
    const float* __restrict__ q, const float* __restrict__ planes,
    unsigned* __restrict__ bitpack, unsigned* __restrict__ cent)
{
  __shared__ float pl[BITS * (EE + 1)];            // 2080 f32
  const int tid = threadIdx.x;
  for (int i = tid; i < BITS * (EE + 1); i += 256) pl[i] = planes[i];
  __syncthreads();

  const int idx = blockIdx.x * 256 + tid;          // [0, NH*LQ)
  const int l  = idx & (LQ - 1);
  const int nh = idx >> 12;
  const int h  = nh & (HH - 1);
  const int n  = nh >> 4;

  const float* __restrict__ qrow = q + (((size_t)n * LQ + l) * HH + h) * EE;
  float qr[EE];
  #pragma unroll
  for (int j = 0; j < EE / 4; ++j) {
    float4 v = *(const float4*)(qrow + 4 * j);
    qr[4 * j + 0] = v.x; qr[4 * j + 1] = v.y;
    qr[4 * j + 2] = v.z; qr[4 * j + 3] = v.w;
  }

  unsigned bits = 0u;
  for (int b = 0; b < BITS; ++b) {
    const float* p = pl + b * (EE + 1);
    float acc = p[EE];                              // bias
    #pragma unroll
    for (int e = 0; e < EE; ++e) acc = fmaf(qr[e], p[e], acc);
    bits |= (acc > 0.0f ? 1u : 0u) << b;
  }
  bitpack[(size_t)nh * LQ + l] = bits;
  if (l < CL) cent[nh * CL + l] = bits;
}

// ---------------------------------------------------------------------------
// K2: one Lloyd step (assign via popcount-argmin, then centroid majority
// update).  final!=0: record assignment + per-cluster counts, skip update.
// sim = 32 - 2*popc(x^c): argmax(sim, first wins) == strict argmin(popc).
// ---------------------------------------------------------------------------
__global__ __launch_bounds__(1024) void k_lloyd(
    const unsigned* __restrict__ bitpack, unsigned* __restrict__ cent,
    int* __restrict__ assign, int* __restrict__ counts, int final_pass)
{
  __shared__ unsigned sc[CL];
  __shared__ unsigned ones[CL * BITS];              // 32 KB
  __shared__ unsigned cnt[CL];
  const int nh  = blockIdx.x;
  const int tid = threadIdx.x;

  for (int i = tid; i < CL; i += 1024) { sc[i] = cent[nh * CL + i]; cnt[i] = 0u; }
  for (int i = tid; i < CL * BITS; i += 1024) ones[i] = 0u;
  __syncthreads();

  for (int l = tid; l < LQ; l += 1024) {
    const unsigned x = bitpack[(size_t)nh * LQ + l];
    int best = 0;
    int bd   = __popc(x ^ sc[0]);
    for (int c = 1; c < CL; ++c) {
      const int d = __popc(x ^ sc[c]);
      if (d < bd) { bd = d; best = c; }             // first index wins ties
    }
    atomicAdd(&cnt[best], 1u);
    if (final_pass) {
      assign[(size_t)nh * LQ + l] = best;
    } else {
      #pragma unroll
      for (int b = 0; b < BITS; ++b)
        atomicAdd(&ones[best * BITS + b], (x >> b) & 1u);
    }
  }
  __syncthreads();

  if (!final_pass) {
    for (int c = tid; c < CL; c += 1024) {
      const unsigned cc = cnt[c];
      unsigned nb = 0u;
      #pragma unroll
      for (int b = 0; b < BITS; ++b)
        if (2u * ones[c * BITS + b] >= cc) nb |= 1u << b;  // sums>=0 -> +1
      cent[nh * CL + c] = nb;
    }
  } else {
    for (int c = tid; c < CL; c += 1024) counts[nh * CL + c] = (int)cnt[c];
  }
}

// ---------------------------------------------------------------------------
// K3: Q_grouped[nh][c][e] = (sum of q rows assigned to c) / max(count,1)
// ---------------------------------------------------------------------------
__global__ __launch_bounds__(1024) void k_qgroup(
    const float* __restrict__ q, const int* __restrict__ assign,
    const int* __restrict__ counts, float* __restrict__ Qg)
{
  extern __shared__ float qs[];                     // CL*EE = 64 KB
  const int nh = blockIdx.x;
  const int n  = nh >> 4;
  const int h  = nh & (HH - 1);
  const int tid  = threadIdx.x;
  const int e    = tid & (EE - 1);
  const int lsub = tid >> 6;                        // 16 rows in parallel

  for (int i = tid; i < CL * EE; i += 1024) qs[i] = 0.0f;
  __syncthreads();

  for (int l = lsub; l < LQ; l += 16) {
    const int c = assign[(size_t)nh * LQ + l];
    const float v = q[(((size_t)n * LQ + l) * HH + h) * EE + e];
    atomicAdd(&qs[c * EE + e], v);
  }
  __syncthreads();

  for (int c = lsub; c < CL; c += 16) {
    const float f = 1.0f / fmaxf((float)counts[nh * CL + c], 1.0f);
    Qg[((size_t)nh * CL + c) * EE + e] = qs[c * EE + e] * f;
  }
}

// ---------------------------------------------------------------------------
// K4: single-pass flash attention: Vc = softmax(temp*Qg*K^T + mask) * V
// grid = NH * 2 c-strips; block = 256 (8 waves); wave -> one 16-row c-tile.
// WMMA bf16 16x16x32, f32 accum.  Fragment maps per CDNA5 ISA tables:
//   A (16x32): lane g*16+m holds row m, elem i -> K = (i/8)*16 + g*8 + i%8
//   B (32x16): lane g*16+n holds col n, elem i -> K = (i/8)*16 + g*8 + i%8
//   D (16x16): lane (m/8)*16+n, vgpr r -> row m = r + 8g, col n = lane%16
// K and V are converted to bf16 ONCE per block (staging / transpose passes)
// and stored as packed bf16-pair dwords so fragment builds are pure ds loads.
// ---------------------------------------------------------------------------
__global__ __launch_bounds__(256) void k_attn(
    const float* __restrict__ Qg, const float* __restrict__ keys,
    const float* __restrict__ values, const int* __restrict__ key_lengths,
    float* __restrict__ Vc)
{
  extern __shared__ char smem_raw[];
  unsigned* Ks16 = (unsigned*)smem_raw;             // [64 s][KST]  bf16 pairs along e
  float*    Vsf  = (float*)(Ks16 + 64 * KST);       // [64 s][PADKV] f32 staging
  unsigned* Vt16 = (unsigned*)(Vsf + 64 * PADKV);   // [64 e][KST]  bf16 pairs along s
  unsigned short* Pb = (unsigned short*)(Vt16 + 64 * KST); // [8][16][PBST]

  const int bid   = blockIdx.x;
  const int strip = bid & 1;
  const int nh    = bid >> 1;
  const int n  = nh >> 4;
  const int h  = nh & (HH - 1);
  const int tid  = threadIdx.x;
  const int wave = tid >> 5;
  const int lane = tid & 31;
  const int lm   = lane & 15;
  const int g    = lane >> 4;

  unsigned short* Pw = Pb + wave * 16 * PBST;
  const int   keylen = key_lengths[n];
  const float temp   = 0.125f;                      // 1/sqrt(E)

  // ---- Q A-fragments (this wave's 16 c-rows), loaded once -------------
  const int crow = strip * 128 + wave * 16 + lm;
  const float* __restrict__ qrow = Qg + ((size_t)nh * CL + crow) * EE;
  v16bf qa[2];
  #pragma unroll
  for (int ch = 0; ch < 2; ++ch) {
    BF16x16 t;
    #pragma unroll
    for (int j = 0; j < 8; ++j) {
      const int e = ch * 32 + (j >> 2) * 16 + g * 8 + 2 * (j & 3);
      t.d[j] = pk_bf16(qrow[e], qrow[e + 1]);
    }
    qa[ch] = t.v;
  }

  float mrow[8], lrow[8];
  v8f acc[4];
  #pragma unroll
  for (int r = 0; r < 8; ++r) { mrow[r] = -3.0e38f; lrow[r] = 0.0f; }
  #pragma unroll
  for (int et = 0; et < 4; ++et)
    #pragma unroll
    for (int r = 0; r < 8; ++r) acc[et][r] = 0.0f;

  const size_t rstr = (size_t)HH * EE;              // float stride between s rows
  const float* __restrict__ Kb = keys   + ((size_t)n * SK * HH + h) * EE;
  const float* __restrict__ Vb = values + ((size_t)n * SK * HH + h) * EE;

  for (int s0 = 0; s0 < SK; s0 += 64) {
    __syncthreads();                                // prev-iter LDS reads done
    // ---- stage: K -> bf16 pairs (along e); V -> f32 -----------------
    for (int i = tid; i < 64 * 16; i += 256) {
      const int row = i >> 4, c4 = (i & 15) * 4;
      const size_t go = (size_t)(s0 + row) * rstr + c4;
      float4 kq = *(const float4*)(Kb + go);
      float4 vq = *(const float4*)(Vb + go);
      unsigned* dk = Ks16 + row * KST + (c4 >> 1);
      dk[0] = pk_bf16(kq.x, kq.y);
      dk[1] = pk_bf16(kq.z, kq.w);
      float* dv = Vsf + row * PADKV + c4;
      dv[0] = vq.x; dv[1] = vq.y; dv[2] = vq.z; dv[3] = vq.w;
      if (s0 + 64 < SK) {                           // lookahead prefetch
        __builtin_prefetch(Kb + go + 64 * rstr, 0, 0);
        __builtin_prefetch(Vb + go + 64 * rstr, 0, 0);
      }
    }
    __syncthreads();

    // ---- transpose V -> bf16 pairs along s (published by P-barrier) --
    for (int i = tid; i < 64 * 32; i += 256) {
      const int e = i >> 5, sp = i & 31;
      Vt16[e * KST + sp] =
          pk_bf16(Vsf[(2 * sp) * PADKV + e], Vsf[(2 * sp + 1) * PADKV + e]);
    }

    // ---- QK^T: 4 sub-tiles of 16 keys, 2 k-chunks each --------------
    v8f d[4];
    #pragma unroll
    for (int st = 0; st < 4; ++st) {
      const unsigned* krow = Ks16 + (st * 16 + lm) * KST;
      v16bf kb[2];
      #pragma unroll
      for (int ch = 0; ch < 2; ++ch) {
        BF16x16 t;
        #pragma unroll
        for (int j = 0; j < 8; ++j) t.d[j] = krow[ch * 16 + (j >> 2) * 8 + g * 4 + (j & 3)];
        kb[ch] = t.v;
      }
      v8f z;
      #pragma unroll
      for (int r = 0; r < 8; ++r) z[r] = 0.0f;
      z     = wmma_bf16f32(qa[0], kb[0], z);
      d[st] = wmma_bf16f32(qa[1], kb[1], z);
    }

    // ---- mask + scale + tile row-max --------------------------------
    float tmax[8];
    #pragma unroll
    for (int r = 0; r < 8; ++r) tmax[r] = -INFINITY;
    #pragma unroll
    for (int st = 0; st < 4; ++st) {
      const bool ok = (s0 + st * 16 + lm) < keylen;
      #pragma unroll
      for (int r = 0; r < 8; ++r) {
        const float x = ok ? d[st][r] * temp : -INFINITY;
        d[st][r] = x;
        tmax[r]  = fmaxf(tmax[r], x);
      }
    }
    #pragma unroll
    for (int off = 1; off < 16; off <<= 1)
      #pragma unroll
      for (int r = 0; r < 8; ++r)
        tmax[r] = fmaxf(tmax[r], __shfl_xor(tmax[r], off));

    // ---- online softmax update --------------------------------------
    float psum[8];
    #pragma unroll
    for (int r = 0; r < 8; ++r) {
      const float nm = fmaxf(mrow[r], tmax[r]);
      const float cs = __expf(mrow[r] - nm);
      mrow[r] = nm;
      lrow[r] *= cs;
      #pragma unroll
      for (int et = 0; et < 4; ++et) acc[et][r] *= cs;
      psum[r] = 0.0f;
    }
    #pragma unroll
    for (int st = 0; st < 4; ++st)
      #pragma unroll
      for (int r = 0; r < 8; ++r) {
        const float p = __expf(d[st][r] - mrow[r]);
        d[st][r] = p;
        psum[r] += p;
      }
    #pragma unroll
    for (int off = 1; off < 16; off <<= 1)
      #pragma unroll
      for (int r = 0; r < 8; ++r) psum[r] += __shfl_xor(psum[r], off);
    #pragma unroll
    for (int r = 0; r < 8; ++r) lrow[r] += psum[r];

    // ---- P (D layout) -> bf16 A layout through LDS ------------------
    #pragma unroll
    for (int st = 0; st < 4; ++st)
      #pragma unroll
      for (int r = 0; r < 8; ++r)
        Pw[(r + 8 * g) * PBST + st * 16 + lm] = f2bf(d[st][r]);
    __syncthreads();                                // publishes P and Vt16

    v16bf pa[2];
    #pragma unroll
    for (int ch = 0; ch < 2; ++ch) {
      BF16x16 t;
      const unsigned short* prow = Pw + lm * PBST;
      #pragma unroll
      for (int j = 0; j < 8; ++j) {
        const int sc = ch * 32 + (j >> 2) * 16 + g * 8 + 2 * (j & 3);
        t.d[j] = *(const unsigned*)(prow + sc);
      }
      pa[ch] = t.v;
    }

    // ---- acc += P * V ----------------------------------------------
    #pragma unroll
    for (int et = 0; et < 4; ++et) {
      const unsigned* vrow = Vt16 + (et * 16 + lm) * KST;
      v16bf vb[2];
      #pragma unroll
      for (int ch = 0; ch < 2; ++ch) {
        BF16x16 t;
        #pragma unroll
        for (int j = 0; j < 8; ++j) t.d[j] = vrow[ch * 16 + (j >> 2) * 8 + g * 4 + (j & 3)];
        vb[ch] = t.v;
      }
      acc[et] = wmma_bf16f32(pa[0], vb[0], acc[et]);
      acc[et] = wmma_bf16f32(pa[1], vb[1], acc[et]);
    }
  }

  // ---- normalize and store Vc --------------------------------------
  #pragma unroll
  for (int r = 0; r < 8; ++r) {
    const float inv = 1.0f / lrow[r];               // keylen>=S/2 => lrow>0
    const int c = strip * 128 + wave * 16 + r + 8 * g;
    float* dst = Vc + ((size_t)nh * CL + c) * EE;
    #pragma unroll
    for (int et = 0; et < 4; ++et) dst[et * 16 + lm] = acc[et][r] * inv;
  }
}

// ---------------------------------------------------------------------------
// K5: out[n,l,h,:] = Vc[nh][assign[nh][l]][:]
// ---------------------------------------------------------------------------
__global__ __launch_bounds__(256) void k_scatter(
    const float* __restrict__ Vc, const int* __restrict__ assign,
    float* __restrict__ out)
{
  const size_t t = (size_t)blockIdx.x * 256 + threadIdx.x;   // float4 index
  const int e4 = (int)(t & 15);
  const size_t r1 = t >> 4;
  const int h = (int)(r1 & (HH - 1));
  const size_t r2 = r1 >> 4;
  const int l = (int)(r2 & (LQ - 1));
  const int n = (int)(r2 >> 12);
  const int nh = n * HH + h;
  const int c  = assign[(size_t)nh * LQ + l];
  const float4 v = *(const float4*)(Vc + ((size_t)nh * CL + c) * EE + e4 * 4);
  *(float4*)(out + t * 4) = v;
}

// ---------------------------------------------------------------------------
// Host-side launcher
// ---------------------------------------------------------------------------
extern "C" void kernel_launch(void* const* d_in, const int* in_sizes, int n_in,
                              void* d_out, int out_size, void* d_ws, size_t ws_size,
                              hipStream_t stream) {
  const float* q      = (const float*)d_in[0];   // (N,L,H,E)
  const float* keys   = (const float*)d_in[1];   // (N,S,H,E)
  const float* values = (const float*)d_in[2];   // (N,S,H,E)
  const float* planes = (const float*)d_in[3];   // (BITS,E+1)
  const int*   klen   = (const int*)d_in[4];     // (N,)
  float* out = (float*)d_out;

  char* ws = (char*)d_ws;
  const size_t SZ_BITS = (size_t)NH * LQ * 4;          // 1 MB
  const size_t SZ_CENT = (size_t)NH * CL * 4;          // 64 KB
  const size_t SZ_ASGN = (size_t)NH * LQ * 4;          // 1 MB
  const size_t SZ_CNT  = (size_t)NH * CL * 4;          // 64 KB
  const size_t SZ_QG   = (size_t)NH * CL * EE * 4;     // 4 MB
  unsigned* bitpack = (unsigned*)ws;
  unsigned* cent    = (unsigned*)(ws + SZ_BITS);
  int*      assign  = (int*)(ws + SZ_BITS + SZ_CENT);
  int*      counts  = (int*)(ws + SZ_BITS + SZ_CENT + SZ_ASGN);
  float*    Qg      = (float*)(ws + SZ_BITS + SZ_CENT + SZ_ASGN + SZ_CNT);
  float*    Vc      = (float*)(ws + SZ_BITS + SZ_CENT + SZ_ASGN + SZ_CNT + SZ_QG);

  // K1: bits + centroid seed
  k_packbits<<<(NH * LQ) / 256, 256, 0, stream>>>(q, planes, bitpack, cent);

  // K2: Lloyd iterations + final assignment/counts
  for (int it = 0; it < ITERS; ++it)
    k_lloyd<<<NH, 1024, 0, stream>>>(bitpack, cent, assign, counts, 0);
  k_lloyd<<<NH, 1024, 0, stream>>>(bitpack, cent, assign, counts, 1);

  // K3: grouped queries
  k_qgroup<<<NH, 1024, CL * EE * sizeof(float), stream>>>(q, assign, counts, Qg);

  // K4: flash attention on cluster queries (WMMA bf16)
  const size_t smem = (size_t)(64 * KST) * 4            // Ks16
                    + (size_t)(64 * PADKV) * 4          // Vsf
                    + (size_t)(64 * KST) * 4            // Vt16
                    + (size_t)(8 * 16 * PBST) * 2;      // P    => 53504 B
  k_attn<<<NH * 2, 256, smem, stream>>>(Qg, keys, values, klen, Vc);

  // K5: scatter to output
  k_scatter<<<(NB * LQ * HH * (EE / 4)) / 256, 256, 0, stream>>>(Vc, assign, out);
}
